// MultiHeadAttention_78950088835652
// MI455X (gfx1250) — matrix operand order repaired and checked by reading, not compiled, not run on hardware
//
#include <hip/hip_runtime.h>

#define B_  4
#define T_  2048
#define C_  1024
#define H_  16
#define HD_ 64
#define BT_ (B_*T_)      // 8192
#define N3C (3*C_)       // 3072

typedef __attribute__((ext_vector_type(16))) __bf16 v16bf;
typedef __attribute__((ext_vector_type(8)))  float  v8f;
typedef int      v4i_vs __attribute__((vector_size(16)));
typedef unsigned u32x4  __attribute__((vector_size(16)));
typedef int      i32x8  __attribute__((vector_size(32)));
typedef int      i32x4  __attribute__((vector_size(16)));

union Frag16 { uint4 q[2]; v16bf v; unsigned u[8]; };

#define AS1 __attribute__((address_space(1)))
#define AS3 __attribute__((address_space(3)))

// ---- feature probes (reported on stderr) ----------------------------------
#if __has_builtin(__builtin_amdgcn_global_load_async_to_lds_b128)
#define HAVE_ASYNC 1
#warning CDNA5 probe: global_load_async_to_lds_b128 AVAILABLE
#else
#define HAVE_ASYNC 0
#warning CDNA5 probe: global_load_async_to_lds_b128 MISSING (vgpr-bounce fallback)
#endif

#if __has_builtin(__builtin_amdgcn_tensor_load_to_lds)
#define HAVE_TDM 1
#warning CDNA5 probe: tensor_load_to_lds AVAILABLE (using 6-arg form)
#else
#define HAVE_TDM 0
#warning CDNA5 probe: tensor_load_to_lds MISSING
#endif

// ---- helpers ---------------------------------------------------------------
__device__ __forceinline__ unsigned pk2(float a, float b) {
  unsigned ua = __float_as_uint(a);
  ua += 0x7FFFu + ((ua >> 16) & 1u);       // RNE
  unsigned ub = __float_as_uint(b);
  ub += 0x7FFFu + ((ub >> 16) & 1u);
  return (ua >> 16) | (ub & 0xFFFF0000u);
}
__device__ __forceinline__ unsigned short f2bf(float f) {
  unsigned u = __float_as_uint(f);
  u += 0x7FFFu + ((u >> 16) & 1u);
  return (unsigned short)(u >> 16);
}
__device__ __forceinline__ v8f wmma_bf16(v16bf a, v16bf b, v8f c) {
  return __builtin_amdgcn_wmma_f32_16x16x32_bf16(false, a, false, b, (short)0, c,
                                                 false, false);
}
__device__ __forceinline__ void async_b128(const void* g, void* l) {
#if HAVE_ASYNC
  __builtin_amdgcn_global_load_async_to_lds_b128(
      (AS1 v4i_vs*)(uintptr_t)g, (AS3 v4i_vs*)(uintptr_t)l, 0, 0);
#endif
}
__device__ __forceinline__ void wait_async0() {
#if __has_builtin(__builtin_amdgcn_s_wait_asynccnt)
  __builtin_amdgcn_s_wait_asynccnt(0);
#else
  asm volatile("s_wait_asynccnt 0x0" ::: "memory");
#endif
}
__device__ __forceinline__ void wait_tensor0() {
#if __has_builtin(__builtin_amdgcn_s_wait_tensorcnt)
  __builtin_amdgcn_s_wait_tensorcnt(0);
#else
  asm volatile("s_wait_tensorcnt 0x0" ::: "memory");
#endif
}

// TDM 2D tile load: rows x cols elements of 2 bytes, contiguous tile
// (tensor == tile), per CDNA5 ISA D# layout.
__device__ __forceinline__ void tdm_load_tile_2d(
    const void* gaddr, void* laddr, unsigned rows, unsigned cols,
    unsigned row_stride_elems) {
#if HAVE_TDM
  const unsigned long long ga = (unsigned long long)(uintptr_t)gaddr;
  const unsigned la = (unsigned)(uintptr_t)laddr;
  u32x4 g0;
  g0[0] = 1u;                                              // count=1, user D#
  g0[1] = la;                                              // lds_addr[31:0]
  g0[2] = (unsigned)(ga & 0xFFFFFFFFu);                    // global_addr[31:0]
  g0[3] = (unsigned)((ga >> 32) & 0x1FFFFFFu) | (2u << 30);// addr[56:32]|type=2
  i32x8 g1;
  g1[0] = 0x00010000;                  // data_size=1 (2B); mask/pad/iter = 0
  g1[1] = (int)(cols << 16);           // tensor_dim0[15:0]  @ bits 63:48
  g1[2] = (int)(rows << 16);           // tensor_dim1[15:0]  @ bits 95:80
  g1[3] = (int)(cols << 16);           // tile_dim0          @ bits 127:112
  g1[4] = (int)rows;                   // tile_dim1          @ bits 143:128
  g1[5] = (int)row_stride_elems;       // tensor_dim0_stride[31:0]
  g1[6] = 0;
  g1[7] = 0;
  i32x4 g2 = {0, 0, 0, 0};
  i32x4 g3 = {0, 0, 0, 0};
  i32x8 g4 = {0, 0, 0, 0, 0, 0, 0, 0};
  __builtin_amdgcn_tensor_load_to_lds(g0, g1, g2, g3, g4, 0);
#endif
}

// ---------------------------------------------------------------------------
// Kernel 1: qkv = x @ W_attn + b_attn, scattered into Q/K/V [B,H,T,64] bf16.
// Block tile 128x128 (8 waves: 4 along M x 2 along N), K-step 32.
// ---------------------------------------------------------------------------
__global__ __launch_bounds__(256) void qkv_gemm_kernel(
    const float* __restrict__ x, const float* __restrict__ W,
    const float* __restrict__ bias,
    unsigned short* __restrict__ Qb, unsigned short* __restrict__ Kb,
    unsigned short* __restrict__ Vb)
{
  __shared__ unsigned short As[128 * 32];   // [m][k] row-major bf16
  __shared__ unsigned short Bs[128 * 32];   // [n][k] col-major bf16

  const int tid  = threadIdx.x;
  const int lane = tid & 31;
  const int wave = tid >> 5;
  const int wm   = wave & 3;
  const int wn   = wave >> 2;
  const int l16  = lane & 15;
  const int hi   = lane >> 4;
  const int mblk = blockIdx.x * 128;
  const int nblk = blockIdx.y * 128;

  const int arow  = tid >> 1;
  const int ahalf = (tid & 1) * 16;
  const int bnl   = tid & 127;
  const int bkh   = (tid >> 7) * 16;

  const v8f zero = {0,0,0,0,0,0,0,0};
  v8f acc[2][4];
  for (int i = 0; i < 2; i++)
    for (int j = 0; j < 4; j++) acc[i][j] = zero;

  for (int k0 = 0; k0 < C_; k0 += 32) {
    { // stage A (f32 -> bf16 with packed converts)
      const float4* src = (const float4*)(x + (size_t)(mblk + arow) * C_ + k0 + ahalf);
      float4 f0 = src[0], f1 = src[1], f2 = src[2], f3 = src[3];
      uint4 o0, o1;
      o0.x = pk2(f0.x, f0.y); o0.y = pk2(f0.z, f0.w);
      o0.z = pk2(f1.x, f1.y); o0.w = pk2(f1.z, f1.w);
      o1.x = pk2(f2.x, f2.y); o1.y = pk2(f2.z, f2.w);
      o1.z = pk2(f3.x, f3.y); o1.w = pk2(f3.z, f3.w);
      uint4* dst = (uint4*)&As[arow * 32 + ahalf];
      dst[0] = o0; dst[1] = o1;
    }
    { // stage B column-major; pack k-pairs straight to dwords
      const float* src = W + (size_t)(k0 + bkh) * N3C + nblk + bnl;
      uint4 o0, o1;
      o0.x = pk2(src[0 * (size_t)N3C], src[1 * (size_t)N3C]);
      o0.y = pk2(src[2 * (size_t)N3C], src[3 * (size_t)N3C]);
      o0.z = pk2(src[4 * (size_t)N3C], src[5 * (size_t)N3C]);
      o0.w = pk2(src[6 * (size_t)N3C], src[7 * (size_t)N3C]);
      o1.x = pk2(src[8 * (size_t)N3C], src[9 * (size_t)N3C]);
      o1.y = pk2(src[10 * (size_t)N3C], src[11 * (size_t)N3C]);
      o1.z = pk2(src[12 * (size_t)N3C], src[13 * (size_t)N3C]);
      o1.w = pk2(src[14 * (size_t)N3C], src[15 * (size_t)N3C]);
      uint4* dst = (uint4*)&Bs[bnl * 32 + bkh];
      dst[0] = o0; dst[1] = o1;
    }
    __syncthreads();

    Frag16 af[2], bf[4];
    const int kb8  = hi * 8;
    const int koff = hi * 16;
    #pragma unroll
    for (int am = 0; am < 2; am++) {
      const int m = wm * 32 + am * 16 + l16;
      af[am].q[0] = *(const uint4*)&As[m * 32 + kb8];
      af[am].q[1] = *(const uint4*)&As[m * 32 + 16 + kb8];
    }
    #pragma unroll
    for (int bn = 0; bn < 4; bn++) {
      const int n = wn * 64 + bn * 16 + l16;
      const uint4* p = (const uint4*)&Bs[n * 32 + kb8];
      bf[bn].q[0] = p[0]; bf[bn].q[1] = p[2];
    }
    #pragma unroll
    for (int am = 0; am < 2; am++)
      #pragma unroll
      for (int bn = 0; bn < 4; bn++)
        acc[am][bn] = wmma_bf16(af[am].v, bf[bn].v, acc[am][bn]);
    __syncthreads();
  }

  // epilogue: +bias, scatter to Q/K/V [B,H,T,64] bf16
  #pragma unroll
  for (int bn = 0; bn < 4; bn++) {
    const int ng = nblk + wn * 64 + bn * 16 + l16;
    const float bv = bias[ng];
    const int which = ng >> 10;
    const int cc = ng & 1023;
    const int h = cc >> 6, d = cc & 63;
    unsigned short* dst = (which == 0) ? Qb : (which == 1) ? Kb : Vb;
    #pragma unroll
    for (int am = 0; am < 2; am++)
      #pragma unroll
      for (int j = 0; j < 8; j++) {
        const int r = mblk + wm * 32 + am * 16 + j + hi * 8;
        const int bb = r >> 11, t = r & 2047;
        dst[((size_t)(bb * H_ + h) * T_ + t) * HD_ + d] = f2bf(acc[am][bn][j] + bv);
      }
  }
}

// ---------------------------------------------------------------------------
// Kernel 2: causal flash attention. 1 block = (b,h) x 64 queries, 4 waves,
// each wave owns 16 queries. Key blocks of 32. Computes S^T = K*Q^T so that
// softmax stats are lane-local and P converts to the A-fragment in-register.
// K tile staged by the Tensor Data Mover (one D# per tile) when available.
// ---------------------------------------------------------------------------
__global__ __launch_bounds__(128) void flash_attn_kernel(
    const unsigned short* __restrict__ Qb,
    const unsigned short* __restrict__ Kb,
    const unsigned short* __restrict__ Vb,
    unsigned short* __restrict__ Yb)
{
  __shared__ unsigned short Kt[32 * 64];    // K block row-major [key][d]
  __shared__ unsigned short Vt[64 * 32];    // V block transposed [d][key]

  const int tid  = threadIdx.x;
  const int lane = tid & 31;
  const int wave = tid >> 5;
  const int l16  = lane & 15;
  const int hi   = lane >> 4;

  const int bh   = blockIdx.x >> 5;
  const int qblk = blockIdx.x & 31;
  const int b    = bh >> 4, h = bh & 15;
  const int q0   = qblk * 64 + wave * 16;

  const size_t base = (size_t)bh * T_ * HD_;

  // Q^T fragments (B-layout) straight from global: contiguous per lane
  Frag16 qf[2];
  {
    const int q = q0 + l16;
    const int koff = hi * 16;
    #pragma unroll
    for (int f = 0; f < 2; f++) {
      const uint4* p = (const uint4*)(Qb + base + (size_t)q * HD_ + f * 32 + koff);
      qf[f].q[0] = p[0]; qf[f].q[1] = p[1];
    }
  }

  const v8f zero = {0,0,0,0,0,0,0,0};
  v8f o[4];
  for (int f = 0; f < 4; f++) o[f] = zero;
  float mrow = -1e30f, lrow = 0.0f;
  const float scale = 0.125f;               // 1/sqrt(64)

  const int nkb = qblk * 2 + 2;             // uniform across block (causal)
  for (int kblk = 0; kblk < nkb; kblk++) {
    // stage K rows: TDM descriptor load (whole 32x64 tile) if available
#if HAVE_TDM
    if (wave == 0) {
      tdm_load_tile_2d(Kb + base + (size_t)kblk * 32 * HD_, &Kt[0],
                       /*rows=*/32, /*cols=*/64, /*stride=*/64);
    }
#elif HAVE_ASYNC
    {
      const int idx = tid * 16;
      const unsigned short* g = Kb + base + (size_t)kblk * 32 * HD_ + idx;
      async_b128(g,     &Kt[idx]);
      async_b128(g + 8, &Kt[idx + 8]);
    }
#else
    {
      const int idx = tid * 16;
      const uint4* p = (const uint4*)(Kb + base + (size_t)kblk * 32 * HD_ + idx);
      uint4* d = (uint4*)&Kt[idx];
      d[0] = p[0]; d[1] = p[1];
    }
#endif
    { // stage V transposed (explicit scatter; DMA cannot transpose)
      const int key = tid >> 2;
      const int dbase = (tid & 3) * 16;
      const uint4* p = (const uint4*)(Vb + base + (size_t)(kblk * 32 + key) * HD_ + dbase);
      uint4 v0 = p[0], v1 = p[1];
      const unsigned short* vs = (const unsigned short*)&v0;
      #pragma unroll
      for (int i = 0; i < 8; i++) Vt[(dbase + i) * 32 + key] = vs[i];
      vs = (const unsigned short*)&v1;
      #pragma unroll
      for (int i = 0; i < 8; i++) Vt[(dbase + 8 + i) * 32 + key] = vs[i];
    }
#if HAVE_TDM
    if (wave == 0) wait_tensor0();
#elif HAVE_ASYNC
    wait_async0();
#endif
    __syncthreads();

    if (kblk + 1 < nkb) {                   // global_prefetch for next tiles
      __builtin_prefetch(Kb + base + (size_t)(kblk + 1) * 32 * HD_ + tid * 16, 0, 1);
      __builtin_prefetch(Vb + base + (size_t)(kblk + 1) * 32 * HD_ + tid * 16, 0, 1);
    }

    // S^T = K(32x64) @ Q^T(64x16): two 16(key)x16(query) f32 frags
    v8f st[2];
    #pragma unroll
    for (int km = 0; km < 2; km++) {
      Frag16 ka[2];
      const int m = km * 16 + l16;
      const int kb8 = hi * 8;
      #pragma unroll
      for (int f = 0; f < 2; f++) {
        ka[f].q[0] = *(const uint4*)&Kt[m * 64 + f * 32 + 2 * kb8];
        ka[f].q[1] = *(const uint4*)&Kt[m * 64 + f * 32 + 2 * kb8 + 8];
      }
      v8f c = zero;
      c = wmma_bf16(ka[0].v, qf[0].v, c);
      c = wmma_bf16(ka[1].v, qf[1].v, c);
      st[km] = c;
    }

    // scale + causal mask
    const int q_abs = q0 + l16;
    float s[2][8];
    #pragma unroll
    for (int km = 0; km < 2; km++)
      #pragma unroll
      for (int j = 0; j < 8; j++) {
        const int key = kblk * 32 + km * 16 + j + hi * 8;
        s[km][j] = (key <= q_abs) ? st[km][j] * scale : -1e30f;
      }

    // online softmax: stats per query = per lane%16
    float mloc = -1e30f;
    #pragma unroll
    for (int km = 0; km < 2; km++)
      for (int j = 0; j < 8; j++) mloc = fmaxf(mloc, s[km][j]);
    mloc = fmaxf(mloc, __shfl_xor(mloc, 16, 32));
    const float mnew  = fmaxf(mrow, mloc);
    const float alpha = __expf(mrow - mnew);
    float psum = 0.0f;
    float p[2][8];
    #pragma unroll
    for (int km = 0; km < 2; km++)
      for (int j = 0; j < 8; j++) { p[km][j] = __expf(s[km][j] - mnew); psum += p[km][j]; }
    psum += __shfl_xor(psum, 16, 32);
    lrow = lrow * alpha + psum;
    mrow = mnew;

    // P into bf16 A-fragment (16q x 32k): pure register packing
    Frag16 pa;
    #pragma unroll
    for (int v = 0; v < 4; v++) {
      pa.u[v]     = pk2(p[0][2 * v], p[0][2 * v + 1]);
      pa.u[v + 4] = pk2(p[1][2 * v], p[1][2 * v + 1]);
    }

    // rescale O rows by alpha
    #pragma unroll
    for (int j = 0; j < 8; j++) {
      const float af = __shfl(alpha, j + hi * 8, 32);
      #pragma unroll
      for (int f = 0; f < 4; f++) o[f][j] *= af;
    }

    // O += P @ V
    #pragma unroll
    for (int f = 0; f < 4; f++) {
      Frag16 vb;
      const uint4* p4 = (const uint4*)&Vt[(f * 16 + l16) * 32 + hi * 8];
      vb.q[0] = p4[0]; vb.q[1] = p4[2];
      o[f] = wmma_bf16(pa.v, vb.v, o[f]);
    }
    __syncthreads();
  }

  // normalize and store y in [B,T,C] bf16
  #pragma unroll
  for (int j = 0; j < 8; j++) {
    const int qrow = j + hi * 8;
    const float inv = 1.0f / __shfl(lrow, qrow, 32);
    const int q = q0 + qrow;
    #pragma unroll
    for (int f = 0; f < 4; f++)
      Yb[(size_t)(b * T_ + q) * C_ + h * HD_ + f * 16 + l16] = f2bf(o[f][j] * inv);
  }
}

// ---------------------------------------------------------------------------
// Kernel 3: out = y @ W_proj + b_proj (f32 out). A tile is already bf16 ->
// staged with async LDS copies when available.
// ---------------------------------------------------------------------------
__global__ __launch_bounds__(256) void proj_gemm_kernel(
    const unsigned short* __restrict__ Yb, const float* __restrict__ W,
    const float* __restrict__ bias, float* __restrict__ out)
{
  __shared__ unsigned short As[128 * 32];
  __shared__ unsigned short Bs[128 * 32];

  const int tid  = threadIdx.x;
  const int lane = tid & 31;
  const int wave = tid >> 5;
  const int wm   = wave & 3;
  const int wn   = wave >> 2;
  const int l16  = lane & 15;
  const int hi   = lane >> 4;
  const int mblk = blockIdx.x * 128;
  const int nblk = blockIdx.y * 128;

  const int arow  = tid >> 1;
  const int ahalf = (tid & 1) * 16;
  const int bnl   = tid & 127;
  const int bkh   = (tid >> 7) * 16;

  const v8f zero = {0,0,0,0,0,0,0,0};
  v8f acc[2][4];
  for (int i = 0; i < 2; i++)
    for (int j = 0; j < 4; j++) acc[i][j] = zero;

  for (int k0 = 0; k0 < C_; k0 += 32) {
    { // A already bf16: async copy to LDS
      const unsigned short* src = Yb + (size_t)(mblk + arow) * C_ + k0 + ahalf;
      unsigned short* dst = &As[arow * 32 + ahalf];
#if HAVE_ASYNC
      async_b128(src,     dst);
      async_b128(src + 8, dst + 8);
#else
      ((uint4*)dst)[0] = ((const uint4*)src)[0];
      ((uint4*)dst)[1] = ((const uint4*)src)[1];
#endif
    }
    { // stage B column-major with packed converts
      const float* src = W + (size_t)(k0 + bkh) * C_ + nblk + bnl;
      uint4 o0, o1;
      o0.x = pk2(src[0 * (size_t)C_], src[1 * (size_t)C_]);
      o0.y = pk2(src[2 * (size_t)C_], src[3 * (size_t)C_]);
      o0.z = pk2(src[4 * (size_t)C_], src[5 * (size_t)C_]);
      o0.w = pk2(src[6 * (size_t)C_], src[7 * (size_t)C_]);
      o1.x = pk2(src[8 * (size_t)C_], src[9 * (size_t)C_]);
      o1.y = pk2(src[10 * (size_t)C_], src[11 * (size_t)C_]);
      o1.z = pk2(src[12 * (size_t)C_], src[13 * (size_t)C_]);
      o1.w = pk2(src[14 * (size_t)C_], src[15 * (size_t)C_]);
      uint4* dst = (uint4*)&Bs[bnl * 32 + bkh];
      dst[0] = o0; dst[1] = o1;
    }
#if HAVE_ASYNC
    wait_async0();
#endif
    __syncthreads();

    Frag16 af[2], bf[4];
    const int kb8  = hi * 8;
    const int koff = hi * 16;
    #pragma unroll
    for (int am = 0; am < 2; am++) {
      const int m = wm * 32 + am * 16 + l16;
      af[am].q[0] = *(const uint4*)&As[m * 32 + kb8];
      af[am].q[1] = *(const uint4*)&As[m * 32 + 16 + kb8];
    }
    #pragma unroll
    for (int bn = 0; bn < 4; bn++) {
      const int n = wn * 64 + bn * 16 + l16;
      const uint4* p = (const uint4*)&Bs[n * 32 + kb8];
      bf[bn].q[0] = p[0]; bf[bn].q[1] = p[2];
    }
    #pragma unroll
    for (int am = 0; am < 2; am++)
      #pragma unroll
      for (int bn = 0; bn < 4; bn++)
        acc[am][bn] = wmma_bf16(af[am].v, bf[bn].v, acc[am][bn]);
    __syncthreads();
  }

  #pragma unroll
  for (int bn = 0; bn < 4; bn++) {
    const int ng = nblk + wn * 64 + bn * 16 + l16;
    const float bv = bias[ng];
    #pragma unroll
    for (int am = 0; am < 2; am++)
      #pragma unroll
      for (int j = 0; j < 8; j++) {
        const int r = mblk + wm * 32 + am * 16 + j + hi * 8;
        out[(size_t)r * C_ + ng] = acc[am][bn][j] + bv;
      }
  }
}

// ---------------------------------------------------------------------------
extern "C" void kernel_launch(void* const* d_in, const int* in_sizes, int n_in,
                              void* d_out, int out_size, void* d_ws, size_t ws_size,
                              hipStream_t stream) {
  const float* x      = (const float*)d_in[0];
  const float* W_attn = (const float*)d_in[1];
  const float* b_attn = (const float*)d_in[2];
  const float* W_proj = (const float*)d_in[3];
  const float* b_proj = (const float*)d_in[4];
  float* out = (float*)d_out;

  const size_t qkv_elems = (size_t)B_ * H_ * T_ * HD_;   // 8M bf16 each
  unsigned short* Qb = (unsigned short*)d_ws;
  unsigned short* Kb = Qb + qkv_elems;
  unsigned short* Vb = Kb + qkv_elems;
  unsigned short* Yb = Vb + qkv_elems;

  qkv_gemm_kernel<<<dim3(BT_ / 128, N3C / 128), 256, 0, stream>>>(
      x, W_attn, b_attn, Qb, Kb, Vb);
  flash_attn_kernel<<<dim3(B_ * H_ * (T_ / 64)), 128, 0, stream>>>(
      Qb, Kb, Vb, Yb);
  proj_gemm_kernel<<<dim3(BT_ / 128, C_ / 128), 256, 0, stream>>>(
      Yb, W_proj, b_proj, out);
}
